// GAT_21062519619905
// MI455X (gfx1250) — matrix operand order repaired and checked by reading, not compiled, not run on hardware
//
#include <hip/hip_runtime.h>
#include <hip/hip_bf16.h>

typedef __attribute__((ext_vector_type(16))) _Float16 v16h;
typedef __attribute__((ext_vector_type(8)))  float    v8f;

#define SLOPE 0.2f
#define BN_EPS 1e-5f
#define NEG_INF_KEY ((int)0x807FFFFF)   // fkey(-inf)

// ---------------- helpers ----------------
__device__ __forceinline__ int fkey(float f) {
    int i = __float_as_int(f);
    return i ^ ((i >> 31) & 0x7FFFFFFF);     // order-preserving int key
}
__device__ __forceinline__ float fdekey(int k) {
    int i = k ^ ((k >> 31) & 0x7FFFFFFF);    // involution (sign bit preserved)
    return __int_as_float(i);
}
__device__ __forceinline__ float leaky(float x) { return x > 0.f ? x : SLOPE * x; }

// ---------------- fills ----------------
__global__ void fill_f32(float* p, float v, size_t n) {
    size_t i = (size_t)blockIdx.x * blockDim.x + threadIdx.x;
    size_t stride = (size_t)gridDim.x * blockDim.x;
    for (; i < n; i += stride) p[i] = v;
}
__global__ void fill_i32(int* p, int v, size_t n) {
    size_t i = (size_t)blockIdx.x * blockDim.x + threadIdx.x;
    size_t stride = (size_t)gridDim.x * blockDim.x;
    for (; i < n; i += stride) p[i] = v;
}

// ---------------- W pre-swizzle into WMMA B-fragment layout (f16) ----------------
// out[((tile*4 + ks)*32 + lane)*16 + j] = W[k, col], k = ks*32 + (j>>3)*16 + (lane>>4)*8 + (j&7),
// col = tile*16 + (lane&15); pad col >= Mtot with 0. K fixed at 128 (ksteps = 4).
__global__ void pack_w(const float* __restrict__ W, _Float16* __restrict__ out,
                       int Mtot, int Mtiles) {
    int idx = blockIdx.x * blockDim.x + threadIdx.x;
    int total = Mtiles * 4 * 32 * 16;
    if (idx >= total) return;
    int j    = idx & 15;
    int lane = (idx >> 4) & 31;
    int ks   = (idx >> 9) & 3;
    int tile = idx >> 11;
    int k    = ks * 32 + (j >> 3) * 16 + (lane >> 4) * 8 + (j & 7);
    int col  = tile * 16 + (lane & 15);
    float v  = (col < Mtot) ? W[(size_t)k * Mtot + col] : 0.f;
    out[idx] = (_Float16)v;
}

// ---------------- WMMA GEMM: Out[N, outStride] = A[N,128] @ Bpack ----------------
// 1 wave per 16-row tile; A frag per ISA 16-bit 16x32 layout:
//   lane&15 = M row, lane>>4 selects K interleave (+8); j<8 -> k=ks*32+half*8+j,
//   j>=8 -> k=ks*32+16+half*8+(j-8). D: VGPR r -> row half*8+r, lane&15 = col.
__global__ __launch_bounds__(256) void gemm_wmma(
    const float* __restrict__ A, const _Float16* __restrict__ Bp,
    float* __restrict__ Out, int Nrows, int Mtiles, int outStride)
{
    const int K = 128;
    int wave = threadIdx.x >> 5;
    int lane = threadIdx.x & 31;
    int nTiles = Nrows >> 4;
    int rowTile = blockIdx.x * 8 + wave;
    if (rowTile >= nTiles) return;           // wave-uniform; EXEC stays all-ones
    int rowBase = rowTile << 4;
    int half = lane >> 4;
    int mrow = lane & 15;

    const float* arow = A + (size_t)(rowBase + mrow) * K;
    v16h afrag[4];
#pragma unroll
    for (int ks = 0; ks < 4; ++ks) {
        const float* ap = arow + ks * 32 + half * 8;
        float4 f0 = *(const float4*)(ap + 0);
        float4 f1 = *(const float4*)(ap + 4);
        float4 f2 = *(const float4*)(ap + 16);
        float4 f3 = *(const float4*)(ap + 20);
        v16h a;
        a[0]  = (_Float16)f0.x; a[1]  = (_Float16)f0.y; a[2]  = (_Float16)f0.z; a[3]  = (_Float16)f0.w;
        a[4]  = (_Float16)f1.x; a[5]  = (_Float16)f1.y; a[6]  = (_Float16)f1.z; a[7]  = (_Float16)f1.w;
        a[8]  = (_Float16)f2.x; a[9]  = (_Float16)f2.y; a[10] = (_Float16)f2.z; a[11] = (_Float16)f2.w;
        a[12] = (_Float16)f3.x; a[13] = (_Float16)f3.y; a[14] = (_Float16)f3.z; a[15] = (_Float16)f3.w;
        afrag[ks] = a;
    }

    for (int tile = 0; tile < Mtiles; ++tile) {
        v8f acc = {0.f, 0.f, 0.f, 0.f, 0.f, 0.f, 0.f, 0.f};
#pragma unroll
        for (int ks = 0; ks < 4; ++ks) {
            v16h b = *(const v16h*)(Bp + (((size_t)(tile * 4 + ks) * 32 + lane) << 4));
            acc = __builtin_amdgcn_wmma_f32_16x16x32_f16(
                false, afrag[ks], false, b, (short)0, acc, false, false);
        }
#pragma unroll
        for (int r = 0; r < 8; ++r) {
            int orow = rowBase + half * 8 + r;
            Out[(size_t)orow * outStride + tile * 16 + mrow] = acc[r];
        }
    }
}

// ---------------- per-node attention logits ----------------
__global__ void alpha_nodes(const float* __restrict__ xw,
                            const float* __restrict__ a_src, const float* __restrict__ a_dst,
                            float* __restrict__ asrc, float* __restrict__ adst,
                            int N, int C, int stride) {
    int gid = blockIdx.x * blockDim.x + threadIdx.x;
    if (gid >= N * 4) return;
    int n = gid >> 2, h = gid & 3;
    const float* row = xw + (size_t)n * stride + h * C;
    const float* as  = a_src + h * C;
    const float* ad  = a_dst + h * C;
    float s = 0.f, d = 0.f;
    for (int c = 0; c < C; ++c) { float v = row[c]; s += v * as[c]; d += v * ad[c]; }
    asrc[gid] = s;
    adst[gid] = d;
}

// ---------------- edge pipeline (self-loops are edges e >= E) ----------------
__device__ __forceinline__ void edge_sd(const int* ei, int e, int E, int& s, int& d) {
    if (e < E) { s = ei[e]; d = ei[E + e]; } else { s = e - E; d = s; }
}

__global__ void edge_max_kernel(const float* __restrict__ asrc, const float* __restrict__ adst,
                                int* __restrict__ amaxKey, const int* __restrict__ ei,
                                int E, int Nn) {
    int e = blockIdx.x * blockDim.x + threadIdx.x;
    if (e >= E + Nn) return;
    int s, d; edge_sd(ei, e, E, s, d);
#pragma unroll
    for (int h = 0; h < 4; ++h) {
        float raw = leaky(asrc[s * 4 + h] + adst[d * 4 + h]);
        atomicMax(&amaxKey[d * 4 + h], fkey(raw));
    }
}

__global__ void edge_denom_kernel(const float* __restrict__ asrc, const float* __restrict__ adst,
                                  const int* __restrict__ amaxKey, float* __restrict__ denom,
                                  const int* __restrict__ ei, int E, int Nn) {
    int e = blockIdx.x * blockDim.x + threadIdx.x;
    if (e >= E + Nn) return;
    int s, d; edge_sd(ei, e, E, s, d);
#pragma unroll
    for (int h = 0; h < 4; ++h) {
        float raw = leaky(asrc[s * 4 + h] + adst[d * 4 + h]);
        atomicAdd(&denom[d * 4 + h], __expf(raw - fdekey(amaxKey[d * 4 + h])));
    }
}

__global__ void scatter_kernel(const float* __restrict__ xw, float* __restrict__ agg,
                               const float* __restrict__ asrc, const float* __restrict__ adst,
                               const int* __restrict__ amaxKey, const float* __restrict__ denom,
                               const int* __restrict__ ei, int E, int Nn, int C, int stride) {
    int gid = blockIdx.x * blockDim.x + threadIdx.x;
    if (gid >= (E + Nn) * 4) return;
    int e = gid >> 2, h = gid & 3;
    int s, d; edge_sd(ei, e, E, s, d);
    float raw  = leaky(asrc[s * 4 + h] + adst[d * 4 + h]);
    float coef = __expf(raw - fdekey(amaxKey[d * 4 + h])) / (denom[d * 4 + h] + 1e-16f);
    const float* sp = xw + (size_t)s * stride + h * C;
    float*       dp = agg + (size_t)d * stride + h * C;
    for (int c = 0; c < C; ++c) atomicAdd(dp + c, sp[c] * coef);
}

// ---------------- bias + ELU + BN stats (stride 128, 1 thread = 1 channel) ----------------
#define POST_ROWS 512
__global__ __launch_bounds__(128) void post_stats(const float* __restrict__ agg,
                                                  const float* __restrict__ bias,
                                                  float* __restrict__ hpre,
                                                  float* __restrict__ sums, int N) {
    int c = threadIdx.x;                     // 0..127
    int r0 = blockIdx.x * POST_ROWS;
    int r1 = r0 + POST_ROWS; if (r1 > N) r1 = N;
    float bc = bias[c];
    float s = 0.f, s2 = 0.f;
    for (int r = r0; r < r1; ++r) {
        float v = agg[(size_t)r * 128 + c] + bc;
        v = v > 0.f ? v : (__expf(v) - 1.f);  // ELU
        hpre[(size_t)r * 128 + c] = v;
        s += v; s2 += v * v;
    }
    atomicAdd(&sums[c], s);
    atomicAdd(&sums[128 + c], s2);
}

__global__ void bn_finalize(const float* __restrict__ sums,
                            const float* __restrict__ gamma, const float* __restrict__ beta,
                            float* __restrict__ scsh, int N) {
    int c = threadIdx.x; if (c >= 128) return;
    float invN = 1.f / (float)N;
    float mean = sums[c] * invN;
    float var  = sums[128 + c] * invN - mean * mean;   // biased
    float inv  = rsqrtf(var + BN_EPS);
    float sc   = gamma[c] * inv;
    scsh[c] = sc;
    scsh[128 + c] = beta[c] - sc * mean;
}

__global__ void bn_apply(float* __restrict__ h, const float* __restrict__ scsh, int N) {
    size_t i = (size_t)blockIdx.x * blockDim.x + threadIdx.x;
    size_t n = (size_t)N * 128;
    size_t stride = (size_t)gridDim.x * blockDim.x;
    for (; i < n; i += stride) {
        int c = (int)(i & 127);
        h[i] = h[i] * scsh[c] + scsh[128 + c];
    }
}

// ---------------- layer-3 head mean + bias ----------------
__global__ void final_out(const float* __restrict__ agg3, const float* __restrict__ b3,
                          float* __restrict__ out, int N) {
    int gid = blockIdx.x * blockDim.x + threadIdx.x;
    if (gid >= N * 2) return;
    int n = gid >> 1, c = gid & 1;
    const float* row = agg3 + (size_t)n * 16;
    float v = 0.25f * (row[c] + row[2 + c] + row[4 + c] + row[6 + c]);
    out[gid] = v + b3[c];
}

// ---------------- launch ----------------
extern "C" void kernel_launch(void* const* d_in, const int* in_sizes, int n_in,
                              void* d_out, int out_size, void* d_ws, size_t ws_size,
                              hipStream_t stream) {
    const float* x   = (const float*)d_in[0];
    const int*   ei  = (const int*)d_in[1];
    const float* W1  = (const float*)d_in[2];
    const float* as1 = (const float*)d_in[3];
    const float* ad1 = (const float*)d_in[4];
    const float* b1  = (const float*)d_in[5];
    const float* g1  = (const float*)d_in[6];
    const float* be1 = (const float*)d_in[7];
    const float* W2  = (const float*)d_in[8];
    const float* as2 = (const float*)d_in[9];
    const float* ad2 = (const float*)d_in[10];
    const float* b2  = (const float*)d_in[11];
    const float* g2  = (const float*)d_in[12];
    const float* be2 = (const float*)d_in[13];
    const float* W3  = (const float*)d_in[14];
    const float* as3 = (const float*)d_in[15];
    const float* ad3 = (const float*)d_in[16];
    const float* b3  = (const float*)d_in[17];
    float* out = (float*)d_out;

    const int N  = in_sizes[0] / 128;
    const int E  = in_sizes[1] / 2;
    const int E2 = E + N;

    // ---- bump allocator over d_ws ----
    size_t off = 0;
    auto alloc = [&](size_t bytes) -> void* {
        void* p = (char*)d_ws + off;
        off += (bytes + 255) & ~(size_t)255;
        return p;
    };
    float*     bufA  = (float*)alloc((size_t)N * 128 * 4);     // activations
    float*     bufB  = (float*)alloc((size_t)N * 128 * 4);     // xw
    float*     bufC  = (float*)alloc((size_t)N * 128 * 4);     // agg
    float*     asrc  = (float*)alloc((size_t)N * 4 * 4);
    float*     adst  = (float*)alloc((size_t)N * 4 * 4);
    int*       amaxK = (int*)  alloc((size_t)N * 4 * 4);
    float*     denom = (float*)alloc((size_t)N * 4 * 4);
    float*     sums  = (float*)alloc(256 * 4);
    float*     scsh  = (float*)alloc(256 * 4);
    _Float16*  wp1   = (_Float16*)alloc(8 * 4 * 32 * 16 * 2);
    _Float16*  wp2   = (_Float16*)alloc(8 * 4 * 32 * 16 * 2);
    _Float16*  wp3   = (_Float16*)alloc(1 * 4 * 32 * 16 * 2);
    float*     xw3   = (float*)alloc((size_t)N * 16 * 4);
    float*     agg3  = (float*)alloc((size_t)N * 16 * 4);

    const int TPB = 256;
    const int gEdge   = (E2 + TPB - 1) / TPB;
    const int gEdgeH  = (E2 * 4 + TPB - 1) / TPB;
    const int gNH     = (N * 4 + TPB - 1) / TPB;
    const int gGemm   = ((N / 16) + 7) / 8;
    const int gPost   = (N + POST_ROWS - 1) / POST_ROWS;

    // ---- weight pre-swizzle ----
    pack_w<<<(8 * 4 * 32 * 16 + TPB - 1) / TPB, TPB, 0, stream>>>(W1, wp1, 128, 8);
    pack_w<<<(8 * 4 * 32 * 16 + TPB - 1) / TPB, TPB, 0, stream>>>(W2, wp2, 128, 8);
    pack_w<<<(1 * 4 * 32 * 16 + TPB - 1) / TPB, TPB, 0, stream>>>(W3, wp3, 8, 1);

    // ================= Layer 1 =================
    gemm_wmma<<<gGemm, 256, 0, stream>>>(x, wp1, bufB, N, 8, 128);
    alpha_nodes<<<gNH, TPB, 0, stream>>>(bufB, as1, ad1, asrc, adst, N, 32, 128);
    fill_f32<<<1024, TPB, 0, stream>>>(bufC, 0.f, (size_t)N * 128);
    fill_i32<<<64, TPB, 0, stream>>>(amaxK, NEG_INF_KEY, (size_t)N * 4);
    fill_f32<<<64, TPB, 0, stream>>>(denom, 0.f, (size_t)N * 4);
    fill_f32<<<1, TPB, 0, stream>>>(sums, 0.f, 256);
    edge_max_kernel<<<gEdge, TPB, 0, stream>>>(asrc, adst, amaxK, ei, E, N);
    edge_denom_kernel<<<gEdge, TPB, 0, stream>>>(asrc, adst, amaxK, denom, ei, E, N);
    scatter_kernel<<<gEdgeH, TPB, 0, stream>>>(bufB, bufC, asrc, adst, amaxK, denom, ei, E, N, 32, 128);
    post_stats<<<gPost, 128, 0, stream>>>(bufC, b1, bufA, sums, N);
    bn_finalize<<<1, 128, 0, stream>>>(sums, g1, be1, scsh, N);
    bn_apply<<<1024, TPB, 0, stream>>>(bufA, scsh, N);

    // ================= Layer 2 =================
    gemm_wmma<<<gGemm, 256, 0, stream>>>(bufA, wp2, bufB, N, 8, 128);
    alpha_nodes<<<gNH, TPB, 0, stream>>>(bufB, as2, ad2, asrc, adst, N, 32, 128);
    fill_f32<<<1024, TPB, 0, stream>>>(bufC, 0.f, (size_t)N * 128);
    fill_i32<<<64, TPB, 0, stream>>>(amaxK, NEG_INF_KEY, (size_t)N * 4);
    fill_f32<<<64, TPB, 0, stream>>>(denom, 0.f, (size_t)N * 4);
    fill_f32<<<1, TPB, 0, stream>>>(sums, 0.f, 256);
    edge_max_kernel<<<gEdge, TPB, 0, stream>>>(asrc, adst, amaxK, ei, E, N);
    edge_denom_kernel<<<gEdge, TPB, 0, stream>>>(asrc, adst, amaxK, denom, ei, E, N);
    scatter_kernel<<<gEdgeH, TPB, 0, stream>>>(bufB, bufC, asrc, adst, amaxK, denom, ei, E, N, 32, 128);
    post_stats<<<gPost, 128, 0, stream>>>(bufC, b2, bufA, sums, N);
    bn_finalize<<<1, 128, 0, stream>>>(sums, g2, be2, scsh, N);
    bn_apply<<<1024, TPB, 0, stream>>>(bufA, scsh, N);

    // ================= Layer 3 =================
    gemm_wmma<<<gGemm, 256, 0, stream>>>(bufA, wp3, xw3, N, 1, 16);
    alpha_nodes<<<gNH, TPB, 0, stream>>>(xw3, as3, ad3, asrc, adst, N, 2, 16);
    fill_f32<<<256, TPB, 0, stream>>>(agg3, 0.f, (size_t)N * 16);
    fill_i32<<<64, TPB, 0, stream>>>(amaxK, NEG_INF_KEY, (size_t)N * 4);
    fill_f32<<<64, TPB, 0, stream>>>(denom, 0.f, (size_t)N * 4);
    edge_max_kernel<<<gEdge, TPB, 0, stream>>>(asrc, adst, amaxK, ei, E, N);
    edge_denom_kernel<<<gEdge, TPB, 0, stream>>>(asrc, adst, amaxK, denom, ei, E, N);
    scatter_kernel<<<gEdgeH, TPB, 0, stream>>>(xw3, agg3, asrc, adst, amaxK, denom, ei, E, N, 2, 16);
    final_out<<<(N * 2 + TPB - 1) / TPB, TPB, 0, stream>>>(agg3, b3, out, N);

    (void)n_in; (void)out_size; (void)ws_size;
}
